// PointNetSetAbstractionMsg_10874857193495
// MI455X (gfx1250) — matrix operand order repaired and checked by reading, not compiled, not run on hardware
//
#include <hip/hip_runtime.h>

// ---------------------------------------------------------------------------
// PointNet++ Set Abstraction (MSG) for MI455X / gfx1250, wave32 + WMMA.
// GEMMs use v_wmma_f32_16x16x32_f16 (f16 inputs, f32 accumulate).
// Activations/weights live in WMMA-fragment-native tiled layouts so the GEMM
// inner loop is pure b128 loads + v_wmma.
// ---------------------------------------------------------------------------

typedef __attribute__((ext_vector_type(16))) _Float16 v16h;
typedef __attribute__((ext_vector_type(8)))  float    v8f;

#define B_    16
#define N_    8192
#define S_    1024
#define OUTC_ 320

// ---------------------------------------------------------------------------
// Kernel 1: farthest point sampling. One block per batch; dist[] in LDS.
// Records fps centroid coords to coord_ce and directly writes out_coord.
// ---------------------------------------------------------------------------
__global__ void fps_kernel(const float* __restrict__ coord, int* __restrict__ fps_idx,
                           float* __restrict__ coord_ce, float* __restrict__ out_coord) {
  __shared__ float dist[N_];
  __shared__ float rv[256];
  __shared__ int   ri[256];
  __shared__ int   cur;
  const int b = blockIdx.x, t = threadIdx.x;
  const float* cx = coord + (size_t)b * 3 * N_;
  const float* cy = cx + N_;
  const float* cz = cy + N_;
  for (int i = t; i < N_; i += 256) dist[i] = 1e10f;
  if (t == 0) cur = 0;
  __syncthreads();
  for (int g = 0; g < S_; ++g) {
    int f = cur;
    float px = cx[f], py = cy[f], pz = cz[f];
    if (t == 0) {
      fps_idx[b * S_ + g] = f;
      float* ce = coord_ce + (size_t)(b * S_ + g) * 3;
      ce[0] = px; ce[1] = py; ce[2] = pz;
      out_coord[(size_t)b * 3 * S_ + 0 * S_ + g] = px;
      out_coord[(size_t)b * 3 * S_ + 1 * S_ + g] = py;
      out_coord[(size_t)b * 3 * S_ + 2 * S_ + g] = pz;
    }
    float bv = -1.0f; int bi = 0;
    for (int i = t; i < N_; i += 256) {
      float dx = cx[i] - px, dy = cy[i] - py, dz = cz[i] - pz;
      float d  = dx * dx + dy * dy + dz * dz;
      float nd = fminf(dist[i], d);
      dist[i] = nd;
      if (nd > bv) { bv = nd; bi = i; }   // strict > keeps smallest index (argmax tie rule)
    }
    rv[t] = bv; ri[t] = bi;
    __syncthreads();
    for (int sfold = 128; sfold > 0; sfold >>= 1) {
      if (t < sfold) {
        float ov = rv[t + sfold]; int oi = ri[t + sfold];
        if (ov > rv[t] || (ov == rv[t] && oi < ri[t])) { rv[t] = ov; ri[t] = oi; }
      }
      __syncthreads();
    }
    if (t == 0) cur = ri[0];
    __syncthreads();
  }
}

// ---------------------------------------------------------------------------
// Kernel 2: ball query for all three radii. One thread per centroid; scan in
// ascending index order == sort-then-first-k semantics of the reference.
// ---------------------------------------------------------------------------
__global__ void ball_kernel(const float* __restrict__ coord, const float* __restrict__ coord_ce,
                            int* __restrict__ g1, int* __restrict__ g2, int* __restrict__ g3) {
  int tid = blockIdx.x * blockDim.x + threadIdx.x;
  if (tid >= B_ * S_) return;
  int b = tid / S_;
  const float* cx = coord + (size_t)b * 3 * N_;
  const float* cy = cx + N_;
  const float* cz = cy + N_;
  const float* ce = coord_ce + (size_t)tid * 3;
  float px = ce[0], py = ce[1], pz = ce[2];
  int* o1 = g1 + (size_t)tid * 16;
  int* o2 = g2 + (size_t)tid * 32;
  int* o3 = g3 + (size_t)tid * 64;
  int c1 = 0, c2 = 0, c3 = 0;
  const float rr1 = 0.1f * 0.1f, rr2 = 0.2f * 0.2f, rr3 = 0.4f * 0.4f;
  for (int j = 0; j < N_; ++j) {
    float dx = cx[j] - px, dy = cy[j] - py, dz = cz[j] - pz;
    float d = dx * dx + dy * dy + dz * dz;
    if (d <= rr3) {
      if (c3 < 64) o3[c3++] = j;
      if (d <= rr2 && c2 < 32) o2[c2++] = j;
      if (d <= rr1 && c1 < 16) o1[c1++] = j;
      if (c1 == 16 && c2 == 32 && c3 == 64) break;
    }
  }
  // backfill with first index (centroid itself guarantees >=1 hit per radius)
  int f1 = o1[0], f2 = o2[0], f3 = o3[0];
  for (int i = c1; i < 16; ++i) o1[i] = f1;
  for (int i = c2; i < 32; ++i) o2[i] = f2;
  for (int i = c3; i < 64; ++i) o3[i] = f3;
}

// ---------------------------------------------------------------------------
// Fragment layouts (CDNA5 ISA 7.12.2, wave32, 16-bit data):
//  A (16xM x 32K):  lane holds row M=lane&15; half h -> K = h + (hi<<4) (this code's contiguous order, not ISA table order)
//  B (32K x 16N):   lane holds col N=lane&15; half h -> K = h + (hi<<4) (this code's contiguous order, not ISA table order)
//  C/D f32 16x16:   v8f element r -> row = r + 8*hi, col = lane&15
// Activations stored fragment-native: xf[((mt*KB + kb)*32 + lane)*16 + h]
// Weights stored fragment-native:     wf[((rt*KB + kb)*32 + lane)*16 + h]
// ---------------------------------------------------------------------------

// Kernel 3: swizzle layer weights (f32 row-major [C_out][C_in]) into A-fragment f16.
__global__ void wfrag_kernel(const float* __restrict__ W, _Float16* __restrict__ wf,
                             int C_out, int C_in, int KB) {
  int id = blockIdx.x * blockDim.x + threadIdx.x;
  int total = (C_out / 16) * KB * 32;
  if (id >= total) return;
  int lane = id & 31;
  int kb   = (id >> 5) % KB;
  int rt   = (id >> 5) / KB;
  int hi   = lane >> 4;
  int o    = rt * 16 + (lane & 15);
  _Float16* dst = wf + (size_t)id * 16;
  for (int h = 0; h < 16; ++h) {
    int k = kb * 32 + (h + (hi << 4));
    float v = (k < C_in) ? W[(size_t)o * C_in + k] : 0.0f;
    dst[h] = (_Float16)v;
  }
}

// Kernel 4: gather neighborhoods into B-fragment-tiled f16 X0 (C_pad=64 rows).
// Channels: 0..31 = feature, 32..34 = relative coords, 35..63 = zero pad.
__global__ void build_x0_kernel(const float* __restrict__ feat, const float* __restrict__ coord,
                                const float* __restrict__ coord_ce, const int* __restrict__ idx,
                                _Float16* __restrict__ xf, int K, int M) {
  int id = blockIdx.x * blockDim.x + threadIdx.x;   // (mt*2 + kb)*32 + lane
  int Mt = M / 16;
  if (id >= Mt * 2 * 32) return;
  int lane = id & 31;
  int kb   = (id >> 5) & 1;
  int mt   = id >> 6;
  int hi   = lane >> 4;
  int col  = mt * 16 + (lane & 15);
  int sk   = S_ * K;
  int b    = col / sk;
  int rem  = col - b * sk;
  int s    = rem / K;
  int kk   = rem - s * K;
  int j    = idx[(size_t)(b * S_ + s) * K + kk];
  const float* ce = coord_ce + (size_t)(b * S_ + s) * 3;
  _Float16 vals[16];
  for (int h = 0; h < 16; ++h) {
    int c = kb * 32 + h + (hi << 4);
    float v;
    if (c < 32)      v = feat[((size_t)b * 32 + c) * N_ + j];
    else if (c < 35) v = coord[((size_t)b * 3 + (c - 32)) * N_ + j] - ce[c - 32];
    else             v = 0.0f;
    vals[h] = (_Float16)v;
  }
  *(v16h*)(xf + (size_t)id * 16) = *(const v16h*)vals;
}

__global__ void zero_stats_kernel(float* sum, float* sumsq) {
  int t = threadIdx.x;
  sum[t] = 0.0f; sumsq[t] = 0.0f;
}

// ---------------------------------------------------------------------------
// Kernel 5: WMMA GEMM. Y[o][m] = sum_k Wf[o][k] * Xf[k][m] + bias[o], plus
// fused per-channel sum / sum-of-squares (shuffle-reduce + f32 global atomics)
// for the batchnorm statistics. Each wave owns one 16-column tile and all
// C_out row tiles; block of 8 waves covers 128 columns.
// ---------------------------------------------------------------------------
template <int KB, int RT>
__global__ void gemm_wmma_kernel(const _Float16* __restrict__ Xf, const _Float16* __restrict__ Wf,
                                 const float* __restrict__ bias, float* __restrict__ Y,
                                 float* __restrict__ sum, float* __restrict__ sumsq, int M) {
  const int lane = threadIdx.x & 31;
  const int wave = threadIdx.x >> 5;
  const int mt   = blockIdx.x * 8 + wave;
  const int hi   = lane >> 4;

  v8f acc[RT];
#pragma unroll
  for (int rt = 0; rt < RT; ++rt) { v8f z = {}; acc[rt] = z; }

  const v16h* xt = (const v16h*)Xf + (size_t)mt * KB * 32 + lane;
  const v16h* wt = (const v16h*)Wf + lane;
  __builtin_prefetch(xt, 0, 0);   // gfx1250 global_prefetch_b8

#pragma unroll
  for (int kb = 0; kb < KB; ++kb) {
    v16h bfrag = xt[kb * 32];
#pragma unroll
    for (int rt = 0; rt < RT; ++rt) {
      v16h afrag = wt[(rt * KB + kb) * 32];
      acc[rt] = __builtin_amdgcn_wmma_f32_16x16x32_f16(
          /*neg_a=*/false, afrag, /*neg_b=*/false, bfrag,
          /*c_mod=*/(short)0, acc[rt], /*reuse_a=*/false, /*reuse_b=*/false);
    }
  }

  const int col = mt * 16 + (lane & 15);
#pragma unroll
  for (int rt = 0; rt < RT; ++rt) {
#pragma unroll
    for (int r = 0; r < 8; ++r) {
      int o = rt * 16 + r + (hi << 3);
      float v = acc[rt][r] + bias[o];
      Y[(size_t)o * M + col] = v;
      // reduce across the 16 lanes that share channel o (xor stays in half-group)
      float s = v, s2 = v * v;
      for (int off = 1; off < 16; off <<= 1) {
        s  += __shfl_xor(s,  off);
        s2 += __shfl_xor(s2, off);
      }
      if ((lane & 15) == 0) {
        atomicAdd(&sum[o],   s);
        atomicAdd(&sumsq[o], s2);
      }
    }
  }
}

// Kernel 6: fold batch statistics into per-channel scale/shift.
__global__ void finalize_stats_kernel(const float* __restrict__ sum, const float* __restrict__ sumsq,
                                      const float* __restrict__ gamma, const float* __restrict__ beta,
                                      float* __restrict__ scale, float* __restrict__ shift,
                                      int C_out, float invM) {
  int o = blockIdx.x * blockDim.x + threadIdx.x;
  if (o >= C_out) return;
  float mean = sum[o] * invM;
  float var  = sumsq[o] * invM - mean * mean;
  float inv  = rsqrtf(var + 1e-5f);
  float sc   = gamma[o] * inv;
  scale[o] = sc;
  shift[o] = beta[o] - mean * sc;
}

// Kernel 7: BN + ReLU, retile into next layer's B-fragment f16 layout.
__global__ void act_tile_kernel(const float* __restrict__ Y, const float* __restrict__ scale,
                                const float* __restrict__ shift, _Float16* __restrict__ xf,
                                int C_out, int KB, int M) {
  int id = blockIdx.x * blockDim.x + threadIdx.x;
  int Mt = M / 16;
  if (id >= Mt * KB * 32) return;
  int lane = id & 31;
  int kb   = (id >> 5) % KB;
  int mt   = (id >> 5) / KB;
  int hi   = lane >> 4;
  int col  = mt * 16 + (lane & 15);
  _Float16 vals[16];
  for (int h = 0; h < 16; ++h) {
    int c = kb * 32 + h + (hi << 4);
    float v = 0.0f;
    if (c < C_out) {
      v = Y[(size_t)c * M + col] * scale[c] + shift[c];
      v = fmaxf(v, 0.0f);
    }
    vals[h] = (_Float16)v;
  }
  *(v16h*)(xf + (size_t)id * 16) = *(const v16h*)vals;
}

// Kernel 8: final layer BN + ReLU + max over k -> out_feature slice.
__global__ void act_maxpool_kernel(const float* __restrict__ Y, const float* __restrict__ scale,
                                   const float* __restrict__ shift, float* __restrict__ outF,
                                   int C_out, int K, int M, int ch_off) {
  int id = blockIdx.x * blockDim.x + threadIdx.x;
  if (id >= B_ * C_out * S_) return;
  int s = id % S_;
  int o = (id / S_) % C_out;
  int b = id / (S_ * C_out);
  size_t base = (size_t)o * M + (size_t)(b * S_ + s) * K;
  float sc = scale[o], sh = shift[o];
  float best = 0.0f;   // relu >= 0
  for (int k = 0; k < K; ++k) {
    float v = Y[base + k] * sc + sh;
    best = fmaxf(best, fmaxf(v, 0.0f));
  }
  outF[(size_t)b * OUTC_ * S_ + (size_t)(ch_off + o) * S_ + s] = best;
}

// ---------------------------------------------------------------------------
// Host orchestration
// ---------------------------------------------------------------------------
static inline void launch_gemm(int KB, int RT, const _Float16* Xf, const _Float16* Wf,
                               const float* bias, float* Y, float* sum, float* sumsq,
                               int M, hipStream_t stream) {
  dim3 grid(M / 128), block(256);
#define GEMM_CASE(kb, rt)                                                              \
  if (KB == kb && RT == rt) {                                                          \
    gemm_wmma_kernel<kb, rt><<<grid, block, 0, stream>>>(Xf, Wf, bias, Y, sum, sumsq, M); \
    return;                                                                            \
  }
  GEMM_CASE(2, 2) GEMM_CASE(1, 2) GEMM_CASE(1, 4) GEMM_CASE(2, 4)
  GEMM_CASE(2, 6) GEMM_CASE(2, 8) GEMM_CASE(3, 8)
#undef GEMM_CASE
}

extern "C" void kernel_launch(void* const* d_in, const int* in_sizes, int n_in,
                              void* d_out, int out_size, void* d_ws, size_t ws_size,
                              hipStream_t stream) {
  (void)in_sizes; (void)n_in; (void)out_size; (void)ws_size;
  const float* feature = (const float*)d_in[0];
  const float* coord   = (const float*)d_in[1];
  auto P = [&](int br, int l, int which) -> const float* {
    return (const float*)d_in[2 + (br * 3 + l) * 4 + which];   // W, b, gamma, beta
  };

  char* ws = (char*)d_ws;
  size_t p = 0;
  auto carve = [&](size_t bytes) -> char* {
    char* r = ws + p;
    p += (bytes + 255) & ~(size_t)255;
    return r;
  };
  int*      fps_idx  = (int*)carve((size_t)B_ * S_ * 4);
  float*    coord_ce = (float*)carve((size_t)B_ * S_ * 3 * 4);
  int*      idx1     = (int*)carve((size_t)B_ * S_ * 16 * 4);
  int*      idx2     = (int*)carve((size_t)B_ * S_ * 32 * 4);
  int*      idx3     = (int*)carve((size_t)B_ * S_ * 64 * 4);
  float*    sum      = (float*)carve(256 * 4);
  float*    sumsq    = (float*)carve(256 * 4);
  float*    scale    = (float*)carve(256 * 4);
  float*    shift    = (float*)carve(256 * 4);
  _Float16* wfrag    = (_Float16*)carve((size_t)8 * 3 * 32 * 16 * 2);
  _Float16* xfrag    = (_Float16*)carve((size_t)96 * 1048576 * 2);   // max C_pad x max M, f16
  float*    Ybuf     = (float*)carve((size_t)128 * 1048576 * 4);     // max C_out x max M, f32

  float* outF = (float*)d_out;
  float* outC = outF + (size_t)B_ * OUTC_ * S_;

  fps_kernel<<<B_, 256, 0, stream>>>(coord, fps_idx, coord_ce, outC);
  ball_kernel<<<(B_ * S_ + 255) / 256, 256, 0, stream>>>(coord, coord_ce, idx1, idx2, idx3);

  const int Ks[3] = {16, 32, 64};
  int* idxs[3] = {idx1, idx2, idx3};
  const int feats[3][4] = {{35, 32, 32, 64}, {35, 64, 64, 128}, {35, 64, 96, 128}};
  const int ch_offs[3] = {0, 64, 192};

  for (int br = 0; br < 3; ++br) {
    const int K = Ks[br];
    const int M = B_ * S_ * K;
    {
      int total = (M / 16) * 2 * 32;
      build_x0_kernel<<<(total + 255) / 256, 256, 0, stream>>>(feature, coord, coord_ce,
                                                               idxs[br], xfrag, K, M);
    }
    for (int l = 0; l < 3; ++l) {
      const int C_in  = feats[br][l];
      const int C_out = feats[br][l + 1];
      const int KB = (C_in + 31) / 32;
      const int RT = C_out / 16;
      {
        int total = RT * KB * 32;
        wfrag_kernel<<<(total + 255) / 256, 256, 0, stream>>>(P(br, l, 0), wfrag, C_out, C_in, KB);
      }
      zero_stats_kernel<<<1, 256, 0, stream>>>(sum, sumsq);
      launch_gemm(KB, RT, xfrag, wfrag, P(br, l, 1), Ybuf, sum, sumsq, M, stream);
      finalize_stats_kernel<<<1, 128, 0, stream>>>(sum, sumsq, P(br, l, 2), P(br, l, 3),
                                                   scale, shift, C_out, 1.0f / (float)M);
      if (l < 2) {
        int KBn = (C_out + 31) / 32;
        int total = (M / 16) * KBn * 32;
        act_tile_kernel<<<(total + 255) / 256, 256, 0, stream>>>(Ybuf, scale, shift, xfrag,
                                                                 C_out, KBn, M);
      } else {
        int total = B_ * C_out * S_;
        act_maxpool_kernel<<<(total + 255) / 256, 256, 0, stream>>>(Ybuf, scale, shift, outF,
                                                                    C_out, K, M, ch_offs[br]);
      }
    }
  }
}
